// DenseMaskedCrossAttention_70068096467025
// MI455X (gfx1250) — compile-verified, matrix operand order checked by reading, not run on hardware
//
#include <hip/hip_runtime.h>
#include <hip/hip_bf16.h>

// Problem constants (match reference)
#define BB 4
#define SS 256
#define PP 1024
#define DD 256
#define HH 8
#define DHH 32   // head dim

typedef __attribute__((ext_vector_type(16))) _Float16 v16h;
typedef __attribute__((ext_vector_type(8)))  _Float16 v8h;
typedef __attribute__((ext_vector_type(8)))  float    v8f;
typedef __attribute__((ext_vector_type(4)))  unsigned int u32x4;
typedef __attribute__((ext_vector_type(8)))  int      i32x8;
typedef __attribute__((ext_vector_type(4)))  int      i32x4;

#if __has_builtin(__builtin_amdgcn_tensor_load_to_lds) && __has_builtin(__builtin_amdgcn_s_wait_tensorcnt)
#define USE_TDM 1
#endif

// ---------------------------------------------------------------------------
// WMMA wrapper: D(16x16,f32) = A(16x32,f16) x B(32x16,f16) + C
// ---------------------------------------------------------------------------
__device__ __forceinline__ v8f wmma_f16(v16h a, v16h b, v8f c) {
    return __builtin_amdgcn_wmma_f32_16x16x32_f16(
        false, a, false, b, (short)0, c, false, false);
}

// 16-bit 16x32 A-tile (and, by lane<->N symmetry, the 32x16 B-tile when each
// lane holds one row of the pre-transposed operand):
//   lane<16 : row = lane,    K elems {0..7, 16..23}
//   lane>=16: row = lane-16, K elems {8..15, 24..31}
template <typename T>
__device__ __forceinline__ v16h load_Atile(const T* base, int stride) {
    unsigned l  = threadIdx.x & 31u;
    int r = l & 15, hi = (l >> 4) & 1;
    const T* row = base + (size_t)r * stride + hi * 8;
    union { v16h v; _Float16 e[16]; } u;
#pragma unroll
    for (int i = 0; i < 8; ++i) {
        u.e[i]     = (_Float16)row[i];
        u.e[i + 8] = (_Float16)row[16 + i];
    }
    return u.v;
}

// ---------------------------------------------------------------------------
// TDM: stage a contiguous run of f16 elements into LDS (1-row 2D tile D#).
// LDS byte offset = low 32 bits of the flat pointer (LDS aperture mapping).
// ---------------------------------------------------------------------------
#ifdef USE_TDM
__device__ __forceinline__ void tdm_copy_to_lds(const _Float16* gsrc, _Float16* ldst,
                                                unsigned nelem) {
    unsigned lds_off = (unsigned)(size_t)ldst;
    unsigned long long ga = (unsigned long long)(size_t)gsrc;
    u32x4 g0;
    g0[0] = 1u;                                            // count=1, user D#
    g0[1] = lds_off;                                       // lds_addr
    g0[2] = (unsigned)(ga & 0xFFFFFFFFu);                  // global_addr lo
    g0[3] = (unsigned)((ga >> 32) & 0x01FFFFFFu) | (2u << 30);  // addr hi | type=2
    i32x8 g1;
    g1[0] = (int)(1u << 16);                               // data_size=1 (2 bytes)
    g1[1] = (int)((nelem & 0xFFFFu) << 16);                // tensor_dim0[15:0]
    g1[2] = (int)(((nelem >> 16) & 0xFFFFu) | (1u << 16)); // tensor_dim0[31:16], tensor_dim1=1
    g1[3] = (int)((nelem & 0xFFFFu) << 16);                // tile_dim0 = nelem
    g1[4] = 1;                                             // tile_dim1=1, tile_dim2=0
    g1[5] = (int)nelem;                                    // tensor_dim0_stride lo
    g1[6] = 0;
    g1[7] = 0;
    i32x4 z4 = {0, 0, 0, 0};
#if defined(__clang_major__) && (__clang_major__ >= 23)
    i32x8 z8 = {0, 0, 0, 0, 0, 0, 0, 0};
    __builtin_amdgcn_tensor_load_to_lds(g0, g1, z4, z4, z8, 0);
#else
    __builtin_amdgcn_tensor_load_to_lds(g0, g1, z4, z4, 0);
#endif
}
#endif

// ---------------------------------------------------------------------------
// Weight packing: fragment layout pack[kt][nt][lane][16] so a consuming wave
// fetches its B-fragment as one contiguous 32B load. (N assumed 256.)
// ---------------------------------------------------------------------------
__global__ void pack_weight_kernel(const float* __restrict__ W,
                                   _Float16* __restrict__ pack, int N) {
    int idx = blockIdx.x * blockDim.x + threadIdx.x;   // (Kd/32)*(N/16)*32*16
    int j = idx & 15;
    int lane = (idx >> 4) & 31;
    int nt = (idx >> 9) & 15;
    int kt = idx >> 13;
    int hi = lane >> 4;
    int n = nt * 16 + (lane & 15);
    int k = kt * 32 + ((j < 8) ? (hi * 8 + j) : (16 + hi * 8 + (j - 8)));
    pack[idx] = (_Float16)W[(size_t)k * N + n];
}

// W1 (16x256, K zero-padded to 32) and W2 (256x8, N zero-padded to 16) packs.
__global__ void pack_mix_kernel(const float* __restrict__ W1,
                                const float* __restrict__ W2,
                                _Float16* __restrict__ W1p,
                                _Float16* __restrict__ W2p) {
    int idx = blockIdx.x * blockDim.x + threadIdx.x;   // 8192 + 4096
    if (idx < 8192) {
        int j = idx & 15, lane = (idx >> 4) & 31, nt = idx >> 9;
        int hi = lane >> 4, n = nt * 16 + (lane & 15);
        _Float16 v = (_Float16)0.f;
        if (j < 8) v = (_Float16)W1[(size_t)(hi * 8 + j) * 256 + n];  // k<16 only
        W1p[idx] = v;
    } else {
        int t = idx - 8192;
        int j = t & 15, lane = (t >> 4) & 31, kt = t >> 9;
        int hi = lane >> 4, n = lane & 15;
        _Float16 v = (_Float16)0.f;
        if (n < 8) {
            int k = kt * 32 + ((j < 8) ? (hi * 8 + j) : (16 + hi * 8 + (j - 8)));
            v = (_Float16)W2[(size_t)k * 8 + n];
        }
        W2p[t] = v;
    }
}

// ---------------------------------------------------------------------------
// K1/K6: Y = X @ W + bias with packed-fragment weights; one wave per 16x16
// tile. MODE 0: row-major OutT store. MODE 1: store V transposed as
// Vt[b,h,dh,p] f16 (for the attn@V B-operand).
// ---------------------------------------------------------------------------
template <typename InT, typename OutT, int MODE>
__global__ void gemm_bias_kernel(const InT* __restrict__ X,
                                 const _Float16* __restrict__ Wpack,
                                 const float* __restrict__ bias,
                                 OutT* __restrict__ Y,
                                 int R, int Kd, int N) {
    int wave = blockIdx.x * (blockDim.x >> 5) + (threadIdx.x >> 5);
    int ntN = N >> 4;
    int tm = wave / ntN, tn = wave % ntN;
    int r0 = tm << 4, n0 = tn << 4;
    unsigned l = threadIdx.x & 31u;

    v8f c = {};
    for (int k0 = 0; k0 < Kd; k0 += 32) {
        v16h a = load_Atile(X + (size_t)r0 * Kd + k0, Kd);
        v16h b = *(const v16h*)(Wpack + (((size_t)(k0 >> 5) * ntN + tn) * 32 + l) * 16);
        c = wmma_f16(a, b, c);
    }
    int n = l & 15, hi = (l >> 4) & 1;
    float bv = bias[n0 + n];
#pragma unroll
    for (int v = 0; v < 8; ++v) {
        int row = r0 + v + hi * 8;
        if constexpr (MODE == 0) {
            Y[(size_t)row * N + n0 + n] = (OutT)(c[v] + bv);
        } else {
            // reshape-to-heads (no transpose) then transpose (p <-> dh)
            int b_ = row >> 10, pb = row & 1023;
            int flat = pb * 256 + (n0 + n);
            int h = flat >> 15, p = (flat >> 5) & 1023, dh = flat & 31;
            Y[(((size_t)b_ * HH + h) * DHH + dh) * PP + p] = (OutT)(c[v] + bv);
        }
    }
}

// ---------------------------------------------------------------------------
// K2: dot[b,h,s,p] = (Q . K) / sqrt(32); one WMMA per 16x16 tile (K = 32)
// ---------------------------------------------------------------------------
__global__ void dot_kernel(const _Float16* __restrict__ Qh,
                           const _Float16* __restrict__ Kh,
                           float* __restrict__ dotb) {
    int wave = blockIdx.x * (blockDim.x >> 5) + (threadIdx.x >> 5);
    int bh = wave >> 10;
    int t  = wave & 1023;
    int ts = t >> 6, tp = t & 63;
    int b = bh >> 3, h = bh & 7;

    const _Float16* qbase = Qh + (size_t)b * SS * DD + (size_t)h * SS * DHH + (size_t)(ts * 16) * DHH;
    const _Float16* kbase = Kh + (size_t)b * PP * DD + (size_t)h * PP * DHH + (size_t)(tp * 16) * DHH;

    v16h a  = load_Atile(qbase, DHH);
    v16h bm = load_Atile(kbase, DHH);
    v8f  c  = {};
    c = wmma_f16(a, bm, c);

    const float scale = 0.17677669529663687f;   // 1/sqrt(32)
    unsigned l = threadIdx.x & 31u;
    int n = l & 15, hi = (l >> 4) & 1;
    size_t base = (((size_t)(b * HH + h)) * SS + ts * 16) * PP + tp * 16 + n;
#pragma unroll
    for (int v = 0; v < 8; ++v)
        dotb[base + (size_t)(v + hi * 8) * PP] = c[v] * scale;
}

// ---------------------------------------------------------------------------
// K3: MixedScoreFF, fused, in-place on the score buffer. W1/W2 fragment packs
// are staged once per block into LDS (TDM if available), then each wave does
// 16 + 8 WMMAs against them.
// ---------------------------------------------------------------------------
__global__ void mix_kernel(float* __restrict__ dotb,
                           const float* __restrict__ mask,
                           const _Float16* __restrict__ W1p,
                           const float* __restrict__ b1,
                           const _Float16* __restrict__ W2p,
                           const float* __restrict__ b2) {
    __shared__ __align__(32) _Float16 w1s[16 * 32 * 16];   // 16 KB
    __shared__ __align__(32) _Float16 w2s[8 * 32 * 16];    // 8 KB
    __shared__ __align__(32) _Float16 hid[4][16 * 256];    // 32 KB (per-wave slab)

#ifdef USE_TDM
    if (threadIdx.x < 32) {                 // wave 0 drives the DMA
        tdm_copy_to_lds(W1p, w1s, 16 * 32 * 16);
        tdm_copy_to_lds(W2p, w2s, 8 * 32 * 16);
    }
    __builtin_amdgcn_s_wait_tensorcnt(0);
    __syncthreads();
#else
    for (int i = threadIdx.x; i < (16 * 32 * 16) / 8; i += blockDim.x)
        *(v8h*)(w1s + i * 8) = *(const v8h*)(W1p + i * 8);
    for (int i = threadIdx.x; i < (8 * 32 * 16) / 8; i += blockDim.x)
        *(v8h*)(w2s + i * 8) = *(const v8h*)(W2p + i * 8);
    __syncthreads();
#endif

    int wslot = threadIdx.x >> 5;
    int wave  = blockIdx.x * (blockDim.x >> 5) + wslot;
    int b   = wave >> 14;
    int rem = wave & 16383;
    int s   = rem >> 6;
    int p0  = (rem & 63) << 4;

    unsigned l = threadIdx.x & 31u;
    int r = l & 15, hi = (l >> 4) & 1;
    int p = p0 + r;

    // A tile: row = (s,p) pair, k = interleaved (dot[h], mask) x8, zero-pad k>=16
    union { v16h v; _Float16 e[16]; } ua;
    size_t dbase = (((size_t)b * HH) * SS + s) * PP + p;
    _Float16 mh = (_Float16)mask[((size_t)b * SS + s) * PP + p];
#pragma unroll
    for (int j = 0; j < 4; ++j) {
        int h = hi * 4 + j;
        ua.e[2 * j]     = (_Float16)dotb[dbase + (size_t)h * SS * PP];
        ua.e[2 * j + 1] = mh;
    }
#pragma unroll
    for (int j = 8; j < 16; ++j) ua.e[j] = (_Float16)0.f;

    int n = r;
    // hidden(16x256) = relu(A @ W1 + b1)
#pragma unroll
    for (int nt = 0; nt < 16; ++nt) {
        v16h ub = *(const v16h*)(w1s + ((size_t)nt * 32 + l) * 16);
        v8f c = {};
        c = wmma_f16(ua.v, ub, c);
        float bb1 = b1[nt * 16 + n];
#pragma unroll
        for (int v = 0; v < 8; ++v) {
            float x = c[v] + bb1;
            hid[wslot][(v + hi * 8) * 256 + nt * 16 + n] = (_Float16)(x > 0.f ? x : 0.f);
        }
    }
    __syncthreads();

    // mixed(16x8) = hidden @ W2, K=256
    v8f c2 = {};
#pragma unroll
    for (int k0 = 0; k0 < 256; k0 += 32) {
        v16h a2 = load_Atile(&hid[wslot][k0], 256);
        v16h ub = *(const v16h*)(w2s + ((size_t)(k0 >> 5) * 32 + l) * 16);
        c2 = wmma_f16(a2, ub, c2);
    }

    // mask -> -inf, overwrite score buffer (disjoint 16x8 slice per wave)
    if (n < 8) {
        size_t mrow = ((size_t)b * SS + s) * PP + p0;
        size_t drow = (((size_t)b * HH + n) * SS + s) * PP + p0;
        float bb2 = b2[n];
#pragma unroll
        for (int v = 0; v < 8; ++v) {
            int m = v + hi * 8;
            float mv  = mask[mrow + m];
            dotb[drow + m] = (mv == 0.f) ? -__builtin_inff() : (c2[v] + bb2);
        }
    }
}

// ---------------------------------------------------------------------------
// K4: softmax over P=1024 per (b,h,s) row; one wave per row, f16 output
// ---------------------------------------------------------------------------
__global__ void softmax_kernel(const float* __restrict__ mixed,
                               _Float16* __restrict__ attn) {
    int wave = blockIdx.x * (blockDim.x >> 5) + (threadIdx.x >> 5);
    unsigned l = threadIdx.x & 31u;
    const float* row = mixed + (size_t)wave * PP + l * 32;

    float v[32];
    float m = -3.402823466e38f;
#pragma unroll
    for (int i = 0; i < 32; ++i) { v[i] = row[i]; m = fmaxf(m, v[i]); }
#pragma unroll
    for (int off = 16; off; off >>= 1) m = fmaxf(m, __shfl_xor(m, off, 32));

    float sum = 0.f;
#pragma unroll
    for (int i = 0; i < 32; ++i) { v[i] = __expf(v[i] - m); sum += v[i]; }
#pragma unroll
    for (int off = 16; off; off >>= 1) sum += __shfl_xor(sum, off, 32);

    float inv = 1.f / sum;
    _Float16* orow = attn + (size_t)wave * PP + l * 32;
#pragma unroll
    for (int i = 0; i < 32; ++i) orow[i] = (_Float16)(v[i] * inv);
}

// ---------------------------------------------------------------------------
// K5: ctx[b,s,h*32+d] = attn[b,h,s,:] @ V[b,h,:,d]; V pre-transposed so the
// B-fragment is a contiguous-row load just like A.
// ---------------------------------------------------------------------------
__global__ void av_kernel(const _Float16* __restrict__ attn,
                          const _Float16* __restrict__ Vt,
                          _Float16* __restrict__ ctx) {
    int wave = blockIdx.x * (blockDim.x >> 5) + (threadIdx.x >> 5);
    int tn = wave & 1;                 // DH/16 = 2 column tiles
    int ts = (wave >> 1) & 15;         // S/16
    int h  = (wave >> 5) & 7;
    int b  = wave >> 8;

    const _Float16* arow = attn + (((size_t)b * HH + h) * SS + ts * 16) * PP;
    const _Float16* vrow = Vt + (((size_t)b * HH + h) * DHH + tn * 16) * PP;

    v8f c = {};
    for (int k0 = 0; k0 < PP; k0 += 32) {
        __builtin_prefetch(arow + k0 + 128, 0, 1);   // global_prefetch_b8
        v16h a  = load_Atile(arow + k0, PP);         // rows = s
        v16h bv = load_Atile(vrow + k0, PP);         // rows = dh (transposed V)
        c = wmma_f16(a, bv, c);
    }
    unsigned l = threadIdx.x & 31u;
    int n = l & 15, hi = (l >> 4) & 1;
#pragma unroll
    for (int v = 0; v < 8; ++v) {
        int s = ts * 16 + v + hi * 8;
        ctx[((size_t)b * SS + s) * DD + h * DHH + tn * 16 + n] = (_Float16)c[v];
    }
}

// ---------------------------------------------------------------------------
// Launch
// ---------------------------------------------------------------------------
extern "C" void kernel_launch(void* const* d_in, const int* in_sizes, int n_in,
                              void* d_out, int out_size, void* d_ws, size_t ws_size,
                              hipStream_t stream) {
    const float* shelf   = (const float*)d_in[0];
    const float* product = (const float*)d_in[1];
    const float* mask    = (const float*)d_in[2];
    const float* Wq = (const float*)d_in[3];  const float* bq = (const float*)d_in[4];
    const float* Wk = (const float*)d_in[5];  const float* bk = (const float*)d_in[6];
    const float* Wv = (const float*)d_in[7];  const float* bv = (const float*)d_in[8];
    const float* Wo = (const float*)d_in[9];  const float* bo = (const float*)d_in[10];
    const float* W1 = (const float*)d_in[11]; const float* b1 = (const float*)d_in[12];
    const float* W2 = (const float*)d_in[13]; const float* b2 = (const float*)d_in[14];
    float* out = (float*)d_out;

    // Workspace layout (~53.5 MB)
    _Float16* Qh  = (_Float16*)d_ws;                       // B*S*D   f16
    _Float16* Kh  = Qh + (size_t)BB * SS * DD;             // B*P*D   f16
    _Float16* Vt  = Kh + (size_t)BB * PP * DD;             // B*H*DH*P f16 (transposed V)
    float*    dotb = (float*)(Vt + (size_t)BB * PP * DD);  // B*H*S*P f32 (in-place mixed)
    _Float16* attn = (_Float16*)(dotb + (size_t)BB * HH * SS * PP);
    _Float16* ctx  = attn + (size_t)BB * HH * SS * PP;     // B*S*D   f16
    _Float16* Wqp = ctx + (size_t)BB * SS * DD;            // 65536 halfs each
    _Float16* Wkp = Wqp + 65536;
    _Float16* Wvp = Wkp + 65536;
    _Float16* Wop = Wvp + 65536;
    _Float16* W1p = Wop + 65536;                           // 8192 halfs
    _Float16* W2p = W1p + 8192;                            // 4096 halfs

    const int TPB = 128;   // 4 waves/block, wave-per-tile everywhere

    // Weight fragment packs
    pack_weight_kernel<<<256, 256, 0, stream>>>(Wq, Wqp, DD);
    pack_weight_kernel<<<256, 256, 0, stream>>>(Wk, Wkp, DD);
    pack_weight_kernel<<<256, 256, 0, stream>>>(Wv, Wvp, DD);
    pack_weight_kernel<<<256, 256, 0, stream>>>(Wo, Wop, DD);
    pack_mix_kernel<<<48, 256, 0, stream>>>(W1, W2, W1p, W2p);

    // Projections
    gemm_bias_kernel<float, _Float16, 0><<<(BB * SS / 16) * (DD / 16) / 4, TPB, 0, stream>>>(
        shelf, Wqp, bq, Qh, BB * SS, DD, DD);
    gemm_bias_kernel<float, _Float16, 0><<<(BB * PP / 16) * (DD / 16) / 4, TPB, 0, stream>>>(
        product, Wkp, bk, Kh, BB * PP, DD, DD);
    gemm_bias_kernel<float, _Float16, 1><<<(BB * PP / 16) * (DD / 16) / 4, TPB, 0, stream>>>(
        product, Wvp, bv, Vt, BB * PP, DD, DD);

    // Scaled QK^T (32768 waves)
    dot_kernel<<<32768 / 4, TPB, 0, stream>>>(Qh, Kh, dotb);

    // Mixed-score FF + mask, in place (65536 waves)
    mix_kernel<<<65536 / 4, TPB, 0, stream>>>(dotb, mask, W1p, b1, W2p, b2);

    // Softmax rows (8192 waves)
    softmax_kernel<<<8192 / 4, TPB, 0, stream>>>(dotb, attn);

    // attn @ V (1024 waves)
    av_kernel<<<1024 / 4, TPB, 0, stream>>>(attn, Vt, ctx);

    // out = ctx @ Wo + bo
    gemm_bias_kernel<_Float16, float, 0><<<(BB * SS / 16) * (DD / 16) / 4, TPB, 0, stream>>>(
        ctx, Wop, bo, out, BB * SS, DD, DD);
}
